// MultiHeadAttention_3350074491061
// MI455X (gfx1250) — matrix operand order repaired and checked by reading, not compiled, not run on hardware
//
#include <hip/hip_runtime.h>
#include <hip/hip_bf16.h>

typedef __bf16 bf16;
typedef __attribute__((ext_vector_type(16))) __bf16 v16bf;
typedef __attribute__((ext_vector_type(8)))  float  v8f;
typedef unsigned int u32x4 __attribute__((ext_vector_type(4)));
typedef int          i32x8 __attribute__((ext_vector_type(8)));
typedef int          i32x4 __attribute__((ext_vector_type(4)));

#define D    128
#define LSEQ 2048
#define H    8
#define R    2
#define BJ   128
#define NJB  (LSEQ / BJ)   // 16

#define LOG2E 1.4426950408889634f
#define LN2   0.6931471805599453f

__device__ __forceinline__ float fast_exp(float x) {
  return __builtin_amdgcn_exp2f(x * LOG2E);
}
__device__ __forceinline__ float fast_log(float x) {
  return LN2 * __builtin_amdgcn_logf(x);
}
__device__ __forceinline__ float fast_pow23(float x) {   // x^(2/3), x>0
  return __builtin_amdgcn_exp2f(0.66666667f * __builtin_amdgcn_logf(x));
}

// ---------------- WMMA fragment helpers (wave32, 16x16x32 bf16) ----------------
// A (16x32): lanes 0-15 -> M=0..15 (K 0..15), lanes 16-31 -> same M, K 16..31.
__device__ __forceinline__ v16bf load_a_rm(const bf16* p, int stride, int k0, int lane) {
  int m = lane & 15;
  int khalf = (lane >> 4) << 4;          // 0 or 16
  const bf16* row = p + m * stride + k0;
  v16bf a;
#pragma unroll
  for (int v = 0; v < 8; ++v) {
    int kb = 2 * v + khalf;
    a[2 * v]     = row[kb];
    a[2 * v + 1] = row[kb + 1];
  }
  return a;
}

// B (32x16), element B[k][n] stored N-major: p[n*stride + k]  (row-major [n][k] source)
__device__ __forceinline__ v16bf load_b_nm(const bf16* p, int stride, int k0, int lane) {
  int n = lane & 15;
  int khalf = (lane >> 4) << 4;          // 0 or 16
  const bf16* col = p + n * stride + k0 + khalf;
  v16bf b;
#pragma unroll
  for (int v = 0; v < 8; ++v) {
    b[2 * v]     = col[2 * v];
    b[2 * v + 1] = col[2 * v + 1];
  }
  return b;
}

// B (32x16), element B[k][n] stored K-major: p[k*stride + n]  (e.g. V tile [k][d])
__device__ __forceinline__ v16bf load_b_km(const bf16* p, int stride, int n0, int lane) {
  int n = lane & 15;
  int khalf = (lane >> 4) << 4;
  const bf16* col = p + khalf * stride + n0 + n;
  v16bf b;
#pragma unroll
  for (int v = 0; v < 8; ++v) {
    b[2 * v]     = col[(2 * v) * stride];
    b[2 * v + 1] = col[(2 * v + 1) * stride];
  }
  return b;
}

__device__ __forceinline__ v8f wmma_bf16(v16bf a, v16bf b, v8f c) {
  return __builtin_amdgcn_wmma_f32_16x16x32_bf16(false, a, false, b, (short)0, c,
                                                 false, false);
}

// ---------------- Tensor Data Mover: contiguous 1-D tile -> LDS ----------------
// n8 = number of 8-byte elements (ISA 8.3/8.4 D# layout; groups 2/3 NULL => <=2D).
// This toolchain's builtin takes 6 args: (g0, g1, g2, g3, g_extra, cpol).
__device__ __forceinline__ void tdm_load_1d(void* lds_dst, const void* gsrc,
                                            unsigned n8) {
  unsigned lds_off = (unsigned)(unsigned long long)lds_dst;   // low 32 bits = LDS addr
  unsigned long long ga = (unsigned long long)gsrc;
  u32x4 g0;
  g0[0] = 1u;                                           // count=1, user descriptor
  g0[1] = lds_off;                                      // lds_addr
  g0[2] = (unsigned)ga;                                 // global_addr[31:0]
  g0[3] = (unsigned)((ga >> 32) & 0x1FFFFFFull) | (2u << 30);  // addr[56:32] | type=2
  i32x8 g1;
  g1[0] = (3 << 16);                                    // data_size = 8 bytes
  g1[1] = (int)((n8 & 0xFFFFu) << 16);                  // tensor_dim0[15:0] @ [63:48]
  g1[2] = (int)(((n8 >> 16) & 0xFFFFu) | (1u << 16));   // tensor_dim0 hi, tensor_dim1=1
  g1[3] = (int)((n8 & 0xFFFFu) << 16);                  // tile_dim0 @ [127:112]
  g1[4] = 0; g1[5] = 0; g1[6] = 0; g1[7] = 0;
  i32x4 z4 = {};
  i32x8 z8 = {};
  __builtin_amdgcn_tensor_load_to_lds(g0, g1, z4, z4, z8, 0);
}

// ---------------- Kernel 0: fp32 -> bf16 staging, cbrt gates ----------------
__global__ void k_convert(const float* q, const float* k, const float* v,
                          const float* src, const float* dest,
                          bf16* qb, bf16* kb, bf16* vb, float* cs, float* cd) {
  long nq = (long)H * R * LSEQ * D;
  long nk = (long)H * LSEQ * D;
  long ns = (long)H * LSEQ;
  long stride = (long)gridDim.x * blockDim.x;
  for (long t = (long)blockIdx.x * blockDim.x + threadIdx.x; t < nq; t += stride) {
    qb[t] = (bf16)q[t];
    if (t < nk) { kb[t] = (bf16)k[t]; vb[t] = (bf16)v[t]; }
    if (t < ns) { cs[t] = cbrtf(src[t]); cd[t] = cbrtf(dest[t]); }
  }
}

// ---------------- Kernel 1: per-row block-boundary prefix carries ----------------
// carry[h][jb][i] = sum_{j' < jb*BJ, j' > i} log1p(-a[i,j'])
__global__ void k_carry(const bf16* kb, const float* cs, const float* cd, float* carry) {
  const int h    = blockIdx.y;
  const int lane = threadIdx.x & 31;
  const int wave = threadIdx.x >> 5;
  const int i0   = (blockIdx.x * 4 + wave) * 16;
  const int lh   = lane >> 4;
  const int N    = lane & 15;
  const bf16*  kh  = kb + (long)h * LSEQ * D;
  const float* csh = cs + h * LSEQ;
  const float* cdh = cd + h * LSEQ;
  float*       ch  = carry + (long)h * NJB * LSEQ;

  v16bf ka[4];
#pragma unroll
  for (int c = 0; c < 4; ++c) ka[c] = load_a_rm(kh + i0 * D, D, 32 * c, lane);

  float csrow[8];
#pragma unroll
  for (int v = 0; v < 8; ++v) csrow[v] = csh[i0 + v + 8 * lh];

  const int jb0 = i0 / BJ;
#pragma unroll
  for (int v = 0; v < 8; ++v) {            // carry into diagonal block is zero
    if (lane == v)      ch[jb0 * LSEQ + i0 + v] = 0.f;
    if (lane == 16 + v) ch[jb0 * LSEQ + i0 + 8 + v] = 0.f;
  }

  float part[8], tot[8];
#pragma unroll
  for (int v = 0; v < 8; ++v) { part[v] = 0.f; tot[v] = 0.f; }

  for (int jc = jb0 * BJ; jc < LSEQ; jc += 16) {
    v8f s2 = {};
#pragma unroll
    for (int c = 0; c < 4; ++c) {
      v16bf bf = load_b_nm(kh + jc * D, D, 32 * c, lane);
      s2 = wmma_bf16(ka[c], bf, s2);
    }
    int   jg  = jc + N;
    float cdj = cdh[jg];
#pragma unroll
    for (int v = 0; v < 8; ++v) {
      int   ig  = i0 + v + 8 * lh;
      float aff = s2[v];
      float t   = fast_pow23(aff) * csrow[v] * cdj;   // NaN for aff<=0, selected away
      float a   = (aff > 0.f) ? t : 0.f;
      a = fminf(a, 0.999999f);
      float la = (jg > ig) ? fast_log(1.f - a) : 0.f;
      part[v] += la;
    }
    if (((jc + 16) & (BJ - 1)) == 0) {       // finished block jc/BJ
#pragma unroll
      for (int v = 0; v < 8; ++v) {          // row-sum across 16 lanes per half
        float s = part[v];
        s += __shfl_xor(s, 1);
        s += __shfl_xor(s, 2);
        s += __shfl_xor(s, 4);
        s += __shfl_xor(s, 8);
        tot[v] += s;
        part[v] = 0.f;
      }
      int jbn = jc / BJ + 1;
      if (jbn < NJB) {
#pragma unroll
        for (int v = 0; v < 8; ++v) {
          if (lane == v)      ch[jbn * LSEQ + i0 + v] = tot[v];
          if (lane == 16 + v) ch[jbn * LSEQ + i0 + 8 + v] = tot[v];
        }
      }
    }
  }
}

// ---------------- Kernel 2: fused flash attention with in-block decay scan ----------------
__global__ void __launch_bounds__(256) k_attn(
    const bf16* kb, const bf16* qb, const bf16* vb,
    const float* cs, const float* cd, const float* carry,
    float* out, float* denom) {
  const int h    = blockIdx.y;
  const int jb   = blockIdx.x;
  const int tid  = threadIdx.x;
  const int lane = tid & 31;
  const int wave = tid >> 5;
  const int lh   = lane >> 4;
  const int N    = lane & 15;
  const int jend = (jb + 1) * BJ;
  const int jt   = jb * BJ + wave * 16;     // this wave's first query column

  __shared__ bf16  ktile[32 * D];
  __shared__ bf16  vtile[32 * D];
  __shared__ float segtot[8][32];
  __shared__ bf16  ptile[8][16 * 32];

  const bf16*  kh  = kb + (long)h * LSEQ * D;
  const bf16*  vh  = vb + (long)h * LSEQ * D;
  const float* csh = cs + h * LSEQ;
  const float* cdh = cd + h * LSEQ;
  const float* ch  = carry + (long)h * NJB * LSEQ;

  // per-wave resident B fragments: K_j columns and Q columns (both r)
  v16bf kjb[4], qfr[R][4];
#pragma unroll
  for (int c = 0; c < 4; ++c) kjb[c] = load_b_nm(kh + jt * D, D, 32 * c, lane);
#pragma unroll
  for (int rr = 0; rr < R; ++rr)
#pragma unroll
    for (int c = 0; c < 4; ++c)
      qfr[rr][c] = load_b_nm(qb + ((long)(h * R + rr) * LSEQ + jt) * D, D, 32 * c, lane);

  const float cdj = cdh[jt + N];            // cbrt(dest_j) for this lane's column

  v8f zero = {};
  v8f acc[R][8];
  float mrun[R], lrun[R];
#pragma unroll
  for (int rr = 0; rr < R; ++rr) {
    mrun[rr] = -1e30f; lrun[rr] = 0.f;
#pragma unroll
    for (int n = 0; n < 8; ++n) acc[rr][n] = zero;
  }

  for (int it = 0; it < jend; it += 32) {
    __syncthreads();
    // stage K and V tiles [32][D] into LDS via the Tensor Data Mover (wave 0 issues)
    if (wave == 0) {
      tdm_load_1d(ktile, kh + it * D, (32 * D * 2) / 8);
      tdm_load_1d(vtile, vh + it * D, (32 * D * 2) / 8);
      __builtin_amdgcn_s_wait_tensorcnt(0);
    }
    if (it + 32 < jend) {
      __builtin_prefetch(kh + (it + 32) * D, 0, 1);
      __builtin_prefetch(vh + (it + 32) * D, 0, 1);
    }
    __syncthreads();

    v16bf ka[2][4];
#pragma unroll
    for (int s = 0; s < 2; ++s)
#pragma unroll
      for (int c = 0; c < 4; ++c)
        ka[s][c] = load_a_rm(ktile + (16 * s) * D, D, 32 * c, lane);

    // ---- decay: aff = K_i . K_j, la = log1p(-a), inclusive scan along j ----
    v8f cum[2];
#pragma unroll
    for (int s = 0; s < 2; ++s) {
      v8f s2 = zero;
#pragma unroll
      for (int c = 0; c < 4; ++c) s2 = wmma_bf16(ka[s][c], kjb[c], s2);
#pragma unroll
      for (int v = 0; v < 8; ++v) {
        int   ig  = it + 16 * s + v + 8 * lh;
        float aff = s2[v];
        float t   = fast_pow23(aff) * csh[ig] * cdj;
        float a   = (aff > 0.f) ? t : 0.f;
        a = fminf(a, 0.999999f);
        float x = ((jt + N) > ig) ? fast_log(1.f - a) : 0.f;
#pragma unroll
        for (int dlt = 1; dlt < 16; dlt <<= 1) {   // 16-lane inclusive scan
          float tt = __shfl_up(x, dlt, 16);
          if (N >= dlt) x += tt;
        }
        cum[s][v] = x;
      }
      if (N == 15) {
#pragma unroll
        for (int v = 0; v < 8; ++v) segtot[wave][16 * s + v + 8 * lh] = cum[s][v];
      }
    }
    __syncthreads();
#pragma unroll
    for (int s = 0; s < 2; ++s) {
#pragma unroll
      for (int v = 0; v < 8; ++v) {
        int   row = 16 * s + v + 8 * lh;
        float off = ch[jb * LSEQ + it + row];       // inter-block carry
        for (int w = 0; w < wave; ++w) off += segtot[w][row];
        cum[s][v] += off;
      }
    }

    // ---- attention update for each r ----
#pragma unroll
    for (int rr = 0; rr < R; ++rr) {
      v8f sc[2];
#pragma unroll
      for (int s = 0; s < 2; ++s) {
        v8f sq = zero;
#pragma unroll
        for (int c = 0; c < 4; ++c) sq = wmma_bf16(ka[s][c], qfr[rr][c], sq);
#pragma unroll
        for (int v = 0; v < 8; ++v) {
          int ig = it + 16 * s + v + 8 * lh;
          sc[s][v] = ((jt + N) >= ig) ? (sq[v] + cum[s][v]) : -1e12f;
        }
      }
      // column-wise (over i) online softmax: reduce over 16 VGPR rows + halves
      float mt = -1e30f;
#pragma unroll
      for (int s = 0; s < 2; ++s)
#pragma unroll
        for (int v = 0; v < 8; ++v) mt = fmaxf(mt, sc[s][v]);
      mt = fmaxf(mt, __shfl_xor(mt, 16));
      float mnew  = fmaxf(mrun[rr], mt);
      float scale = fast_exp(mrun[rr] - mnew);
      float psum  = 0.f;
#pragma unroll
      for (int s = 0; s < 2; ++s)
#pragma unroll
        for (int v = 0; v < 8; ++v) {
          float p = fast_exp(sc[s][v] - mnew);
          sc[s][v] = p;
          psum += p;
        }
      psum += __shfl_xor(psum, 16);
      lrun[rr] = lrun[rr] * scale + psum;
      mrun[rr] = mnew;

      float sr[8];
#pragma unroll
      for (int v = 0; v < 8; ++v) sr[v] = __shfl(scale, v + 8 * lh);
#pragma unroll
      for (int n = 0; n < 8; ++n)
#pragma unroll
        for (int v = 0; v < 8; ++v) acc[rr][n][v] *= sr[v];

      // transpose P through LDS into A layout: ptile[wave][j_local][i_local]
#pragma unroll
      for (int s = 0; s < 2; ++s)
#pragma unroll
        for (int v = 0; v < 8; ++v)
          ptile[wave][N * 32 + 16 * s + v + 8 * lh] = (bf16)sc[s][v];
      v16bf pa = load_a_rm(&ptile[wave][0], 32, 0, lane);
#pragma unroll
      for (int n = 0; n < 8; ++n) {
        v16bf vf = load_b_km(vtile, D, 16 * n, lane);
        acc[rr][n] = wmma_bf16(pa, vf, acc[rr][n]);
      }
    }
  }

  // ---- epilogue: normalize, store out and logsumexp denom ----
#pragma unroll
  for (int rr = 0; rr < R; ++rr) {
    float* oh = out + ((long)(h * R + rr) * LSEQ + jt) * D;
#pragma unroll
    for (int v = 0; v < 8; ++v) {
      float li  = __shfl(lrun[rr], v + 8 * lh);
      float inv = 1.f / li;
      int   row = v + 8 * lh;
#pragma unroll
      for (int n = 0; n < 8; ++n)
        oh[row * D + 16 * n + N] = acc[rr][n][v] * inv;
    }
    if (lane < 16)
      denom[(long)(h * R + rr) * LSEQ + jt + lane] = mrun[rr] + fast_log(lrun[rr]);
  }
}

// ---------------- host launch ----------------
extern "C" void kernel_launch(void* const* d_in, const int* in_sizes, int n_in,
                              void* d_out, int out_size, void* d_ws, size_t ws_size,
                              hipStream_t stream) {
  const float* q    = (const float*)d_in[0];
  const float* k    = (const float*)d_in[1];
  const float* v    = (const float*)d_in[2];
  const float* src  = (const float*)d_in[3];
  const float* dest = (const float*)d_in[4];
  float* out   = (float*)d_out;                     // [h][r][L][D]
  float* denom = out + (long)H * R * LSEQ * D;      // [h][r][L]

  char* ws = (char*)d_ws;
  bf16*  kbf   = (bf16*)ws;  ws += (size_t)H * LSEQ * D * 2;
  bf16*  vbf   = (bf16*)ws;  ws += (size_t)H * LSEQ * D * 2;
  bf16*  qbf   = (bf16*)ws;  ws += (size_t)H * R * LSEQ * D * 2;
  float* carry = (float*)ws; ws += (size_t)H * NJB * LSEQ * 4;
  float* csrc  = (float*)ws; ws += (size_t)H * LSEQ * 4;
  float* cdst  = (float*)ws; ws += (size_t)H * LSEQ * 4;

  k_convert<<<2048, 256, 0, stream>>>(q, k, v, src, dest, qbf, kbf, vbf, csrc, cdst);
  k_carry<<<dim3(LSEQ / 64, H), 128, 0, stream>>>(kbf, csrc, cdst, carry);
  k_attn<<<dim3(NJB, H), 256, 0, stream>>>(kbf, qbf, vbf, csrc, cdst, carry, out, denom);
}
